// ViTRoPEAttention_22969485099420
// MI455X (gfx1250) — matrix operand order repaired and checked by reading, not compiled, not run on hardware
//
#include <hip/hip_runtime.h>
#include <stdint.h>

typedef __attribute__((ext_vector_type(16))) _Float16 v16h;
typedef __attribute__((ext_vector_type(8)))  float    v8f;

#define NHEADS 16
#define HDIM   64
#define SEQ    1024
#define BATCH  16
#define DMODEL 1024
#define MTOT   (BATCH * SEQ)          // 16384 rows

union Frag16 { v16h v; uint32_t u[8]; _Float16 h[16]; };

// ---------------------------------------------------------------- convert
__global__ void cvt_f32_f16(const float* __restrict__ in, _Float16* __restrict__ out, int n) {
  int i = blockIdx.x * blockDim.x + threadIdx.x;
  int stride = gridDim.x * blockDim.x;
  for (; i < n; i += stride) out[i] = (_Float16)in[i];
}

// ---------------------------------------------------------------- QKV GEMM
// y = x @ W.T + b (scaled), f16 out in (b, nh, s, hd) layout.
// Register-blocked: one wave computes a 32x64 tile = 2x4 WMMA subtiles.
__global__ __launch_bounds__(256) void gemm_qkv(
    const _Float16* __restrict__ A,      // (16384, 1024) row-major f16
    const _Float16* __restrict__ W,      // (1024, 1024) row-major f16
    const float* __restrict__ bias,
    _Float16* __restrict__ dst,          // (B, NH, S, HD)
    float postscale) {
  const int lane = threadIdx.x & 31;
  const int wave = threadIdx.x >> 5;
  const int t = blockIdx.x * 8 + wave;      // (16384/32)*(1024/64) = 8192 wave tiles
  const int m0 = (t >> 4) << 5;
  const int n0 = (t & 15) << 6;

  const int half = lane >> 4;
  const int l16  = lane & 15;
  const int akb  = half * 8;                // half the A lane K base (loads use 2*akb = 16*half)
  const int bko  = half * 16;               // B-layout lane K base

  const _Float16* arow[2];
  arow[0] = A + (m0 + l16) * DMODEL;
  arow[1] = A + (m0 + 16 + l16) * DMODEL;
  const _Float16* brow[4];
#pragma unroll
  for (int nt = 0; nt < 4; ++nt) brow[nt] = W + (n0 + nt * 16 + l16) * DMODEL;

  v8f acc[2][4] = {};
  for (int k0 = 0; k0 < DMODEL; k0 += 32) {
    Frag16 a[2], b[4];
#pragma unroll
    for (int mt = 0; mt < 2; ++mt)
#pragma unroll
      for (int j = 0; j < 4; ++j) {
        a[mt].u[j]     = *(const uint32_t*)(arow[mt] + k0 + 2 * akb + 2 * j);
        a[mt].u[4 + j] = *(const uint32_t*)(arow[mt] + k0 + 2 * akb + 8 + 2 * j);
      }
#pragma unroll
    for (int nt = 0; nt < 4; ++nt)
#pragma unroll
      for (int j = 0; j < 8; ++j)
        b[nt].u[j] = *(const uint32_t*)(brow[nt] + k0 + bko + 2 * j);
#pragma unroll
    for (int mt = 0; mt < 2; ++mt)
#pragma unroll
      for (int nt = 0; nt < 4; ++nt)
        acc[mt][nt] = __builtin_amdgcn_wmma_f32_16x16x32_f16(
            false, a[mt].v, false, b[nt].v, (short)0, acc[mt][nt], false, false);
  }

#pragma unroll
  for (int nt = 0; nt < 4; ++nt) {
    const int col = n0 + nt * 16 + l16;
    const float bv = bias[col];
    const int nh = col >> 6, hd = col & 63;
#pragma unroll
    for (int mt = 0; mt < 2; ++mt)
#pragma unroll
      for (int i = 0; i < 8; ++i) {
        int row = m0 + mt * 16 + i + half * 8;   // C layout rows
        int bb = row >> 10, s = row & (SEQ - 1);
        float val = (acc[mt][nt][i] + bv) * postscale;
        dst[(((bb * NHEADS + nh) * SEQ) + s) * HDIM + hd] = (_Float16)val;
      }
  }
}

// ---------------------------------------------------------------- RoPE (in place, pairwise)
__global__ void rope_f16(_Float16* __restrict__ q, const float* __restrict__ cosb,
                         const float* __restrict__ sinb, int npairs) {
  int i = blockIdx.x * blockDim.x + threadIdx.x;
  if (i >= npairs) return;
  int base = i * 2;
  int hd = base & (HDIM - 1);
  int s  = (base >> 6) & (SEQ - 1);
  float x0 = (float)q[base], x1 = (float)q[base + 1];
  float c0 = cosb[s * HDIM + hd], c1 = cosb[s * HDIM + hd + 1];
  float s0 = sinb[s * HDIM + hd], s1 = sinb[s * HDIM + hd + 1];
  q[base]     = (_Float16)(x0 * c0 - x1 * s0);
  q[base + 1] = (_Float16)(x1 * c1 + x0 * s1);
}

// ---------------------------------------------------------------- fused attention
// 8 waves per block share one (b,h); each wave owns a 16-row q block.
// Per 32-key block: K/V panels cooperatively staged in LDS (contiguous 4KB
// each, one uint4 per thread), then all fragment gathers hit LDS banks.
__global__ __launch_bounds__(256) void attn_flash(
    const _Float16* __restrict__ Q, const _Float16* __restrict__ K,
    const _Float16* __restrict__ V, _Float16* __restrict__ O) {
  __shared__ _Float16 Kt[32 * HDIM];        // 4 KB staged K panel
  __shared__ _Float16 Vt[32 * HDIM];        // 4 KB staged V panel
  __shared__ _Float16 ldsP[8][16 * 32];     // 1 KB per wave: C->A relayout of P

  const int tid  = threadIdx.x;
  const int lane = tid & 31;
  const int wave = tid >> 5;
  const int qb = blockIdx.x * 8 + wave;     // 16384 q blocks; qb>>6 uniform per block
  const int bh = qb >> 6;
  const int q0 = (qb & 63) << 4;
  const _Float16* Qh = Q + bh * SEQ * HDIM;
  const _Float16* Kh = K + bh * SEQ * HDIM;
  const _Float16* Vh = V + bh * SEQ * HDIM;

  const int half = lane >> 4;
  const int l16  = lane & 15;

  // Q A-fragments: 16x32 tiles over head-dim (k0 = 0, 32), loaded once
  Frag16 qa[2];
  {
    const _Float16* qrow = Qh + (q0 + l16) * HDIM;
#pragma unroll
    for (int t = 0; t < 2; ++t) {
#pragma unroll
      for (int j = 0; j < 4; ++j) {
        qa[t].u[j]     = *(const uint32_t*)(qrow + t * 32 + half * 16 + 2 * j);
        qa[t].u[4 + j] = *(const uint32_t*)(qrow + t * 32 + half * 16 + 8 + 2 * j);
      }
    }
  }

  v8f o0 = {}, o1 = {}, o2 = {}, o3 = {};
  float mrow[8], lrow[8];
#pragma unroll
  for (int i = 0; i < 8; ++i) { mrow[i] = -3.0e38f; lrow[i] = 0.0f; }

  for (int kb = 0; kb < SEQ; kb += 32) {
    // ---- cooperative stage: K/V 32x64 panels are contiguous 4KB regions
    {
      const uint4* ksrc = (const uint4*)(Kh + (size_t)kb * HDIM);
      const uint4* vsrc = (const uint4*)(Vh + (size_t)kb * HDIM);
      ((uint4*)Kt)[tid] = ksrc[tid];
      ((uint4*)Vt)[tid] = vsrc[tid];
    }
    __syncthreads();

    // ---- scores: two 16x16 key tiles, contraction 64 = 2 WMMAs each
    v8f sc[2] = {};
#pragma unroll
    for (int t = 0; t < 2; ++t) {
      const _Float16* krow = Kt + (t * 16 + l16) * HDIM;
#pragma unroll
      for (int ks = 0; ks < 2; ++ks) {
        Frag16 b;
#pragma unroll
        for (int j = 0; j < 8; ++j)
          b.u[j] = *(const uint32_t*)(krow + ks * 32 + half * 16 + 2 * j);
        sc[t] = __builtin_amdgcn_wmma_f32_16x16x32_f16(false, qa[ks].v, false, b.v,
                                                       (short)0, sc[t], false, false);
      }
    }

    // ---- online softmax (rows live per-VGPR per-half-wave)
    float scale[8], rs[8];
#pragma unroll
    for (int i = 0; i < 8; ++i) {
      float v = fmaxf(sc[0][i], sc[1][i]);
      v = fmaxf(v, __shfl_xor(v, 1, 32));
      v = fmaxf(v, __shfl_xor(v, 2, 32));
      v = fmaxf(v, __shfl_xor(v, 4, 32));
      v = fmaxf(v, __shfl_xor(v, 8, 32));
      float mnew = fmaxf(mrow[i], v);
      scale[i] = __expf(mrow[i] - mnew);
      mrow[i] = mnew;
    }
#pragma unroll
    for (int i = 0; i < 8; ++i) {
      float p0 = __expf(sc[0][i] - mrow[i]);
      float p1 = __expf(sc[1][i] - mrow[i]);
      int row = i + half * 8;
      ldsP[wave][row * 32 + l16]      = (_Float16)p0;
      ldsP[wave][row * 32 + 16 + l16] = (_Float16)p1;
      float r = p0 + p1;
      r += __shfl_xor(r, 1, 32);
      r += __shfl_xor(r, 2, 32);
      r += __shfl_xor(r, 4, 32);
      r += __shfl_xor(r, 8, 32);
      rs[i] = r;
    }
#pragma unroll
    for (int i = 0; i < 8; ++i) {
      lrow[i] = lrow[i] * scale[i] + rs[i];
      o0[i] *= scale[i]; o1[i] *= scale[i]; o2[i] *= scale[i]; o3[i] *= scale[i];
    }

    __syncthreads();                         // P stores visible for A-layout reload
    Frag16 pa;
#pragma unroll
    for (int j = 0; j < 4; ++j) {
      pa.u[j]     = *(const uint32_t*)&ldsP[wave][l16 * 32 + half * 16 + 2 * j];
      pa.u[4 + j] = *(const uint32_t*)&ldsP[wave][l16 * 32 + half * 16 + 8 + 2 * j];
    }

    // ---- O += P @ V  (4 head-dim tiles, contraction = 32 keys; V from LDS)
#pragma unroll
    for (int t = 0; t < 4; ++t) {
      Frag16 vb;
#pragma unroll
      for (int j = 0; j < 8; ++j) {
        int k0 = half * 16 + 2 * j;
        uint32_t lo = *(const uint16_t*)(Vt + k0 * HDIM + t * 16 + l16);
        uint32_t hi = *(const uint16_t*)(Vt + (k0 + 1) * HDIM + t * 16 + l16);
        vb.u[j] = lo | (hi << 16);
      }
      v8f& ot = (t == 0) ? o0 : (t == 1) ? o1 : (t == 2) ? o2 : o3;
      ot = __builtin_amdgcn_wmma_f32_16x16x32_f16(false, pa.v, false, vb.v,
                                                  (short)0, ot, false, false);
    }
    __syncthreads();                         // protect Kt/Vt/ldsP before next stage
  }

  // ---- normalize + store to (b, s, d) layout f16
  const int bb = bh >> 4, nh = bh & 15;
#pragma unroll
  for (int i = 0; i < 8; ++i) {
    float inv = 1.0f / lrow[i];
    int row = q0 + i + half * 8;
    _Float16* orow = O + ((size_t)(bb * SEQ + row)) * DMODEL + nh * HDIM;
    orow[0 * 16 + l16] = (_Float16)(o0[i] * inv);
    orow[1 * 16 + l16] = (_Float16)(o1[i] * inv);
    orow[2 * 16 + l16] = (_Float16)(o2[i] * inv);
    orow[3 * 16 + l16] = (_Float16)(o3[i] * inv);
  }
}

// ---------------------------------------------------------------- output projection
// Same 32x64 register blocking; fp32 output + bias.
__global__ __launch_bounds__(256) void gemm_out(
    const _Float16* __restrict__ A, const _Float16* __restrict__ W,
    const float* __restrict__ bias, float* __restrict__ out) {
  const int lane = threadIdx.x & 31;
  const int wave = threadIdx.x >> 5;
  const int t = blockIdx.x * 8 + wave;      // 8192 wave tiles
  const int m0 = (t >> 4) << 5;
  const int n0 = (t & 15) << 6;

  const int half = lane >> 4;
  const int l16  = lane & 15;
  const int akb  = half * 8;
  const int bko  = half * 16;

  const _Float16* arow[2];
  arow[0] = A + (m0 + l16) * DMODEL;
  arow[1] = A + (m0 + 16 + l16) * DMODEL;
  const _Float16* brow[4];
#pragma unroll
  for (int nt = 0; nt < 4; ++nt) brow[nt] = W + (n0 + nt * 16 + l16) * DMODEL;

  v8f acc[2][4] = {};
  for (int k0 = 0; k0 < DMODEL; k0 += 32) {
    Frag16 a[2], b[4];
#pragma unroll
    for (int mt = 0; mt < 2; ++mt)
#pragma unroll
      for (int j = 0; j < 4; ++j) {
        a[mt].u[j]     = *(const uint32_t*)(arow[mt] + k0 + 2 * akb + 2 * j);
        a[mt].u[4 + j] = *(const uint32_t*)(arow[mt] + k0 + 2 * akb + 8 + 2 * j);
      }
#pragma unroll
    for (int nt = 0; nt < 4; ++nt)
#pragma unroll
      for (int j = 0; j < 8; ++j)
        b[nt].u[j] = *(const uint32_t*)(brow[nt] + k0 + bko + 2 * j);
#pragma unroll
    for (int mt = 0; mt < 2; ++mt)
#pragma unroll
      for (int nt = 0; nt < 4; ++nt)
        acc[mt][nt] = __builtin_amdgcn_wmma_f32_16x16x32_f16(
            false, a[mt].v, false, b[nt].v, (short)0, acc[mt][nt], false, false);
  }

#pragma unroll
  for (int nt = 0; nt < 4; ++nt) {
    const float bv = bias[n0 + nt * 16 + l16];
#pragma unroll
    for (int mt = 0; mt < 2; ++mt)
#pragma unroll
      for (int i = 0; i < 8; ++i) {
        int row = m0 + mt * 16 + i + half * 8;
        out[(size_t)row * DMODEL + n0 + nt * 16 + l16] = acc[mt][nt][i] + bv;
      }
  }
}

// ---------------------------------------------------------------- launch
extern "C" void kernel_launch(void* const* d_in, const int* in_sizes, int n_in,
                              void* d_out, int out_size, void* d_ws, size_t ws_size,
                              hipStream_t stream) {
  const float* x    = (const float*)d_in[0];
  const float* cosb = (const float*)d_in[1];
  const float* sinb = (const float*)d_in[2];
  const float* Wq   = (const float*)d_in[3];
  const float* bq   = (const float*)d_in[4];
  const float* Wk   = (const float*)d_in[5];
  const float* bk   = (const float*)d_in[6];
  const float* Wv   = (const float*)d_in[7];
  const float* bv   = (const float*)d_in[8];
  const float* Wo   = (const float*)d_in[9];
  const float* bo   = (const float*)d_in[10];
  float* out = (float*)d_out;

  char* ws = (char*)d_ws;
  size_t off = 0;
  auto take = [&](size_t bytes) {
    char* p = ws + off;
    off += (bytes + 255) & ~(size_t)255;
    return p;
  };
  const size_t xBytes = (size_t)MTOT * DMODEL * sizeof(_Float16);      // 32 MB
  const size_t wBytes = (size_t)DMODEL * DMODEL * sizeof(_Float16);    // 2 MB
  const size_t qBytes = (size_t)BATCH * NHEADS * SEQ * HDIM * sizeof(_Float16); // 32 MB

  _Float16* x16  = (_Float16*)take(xBytes);
  _Float16* wq16 = (_Float16*)take(wBytes);
  _Float16* wk16 = (_Float16*)take(wBytes);
  _Float16* wv16 = (_Float16*)take(wBytes);
  _Float16* wo16 = (_Float16*)take(wBytes);
  _Float16* q16  = (_Float16*)take(qBytes);
  _Float16* k16  = (_Float16*)take(qBytes);
  _Float16* v16  = (_Float16*)take(qBytes);
  _Float16* a16  = x16;   // x16 dead after QKV GEMMs; reuse for attention output

  // 1) f32 -> f16 conversions
  cvt_f32_f16<<<4096, 256, 0, stream>>>(x, x16, MTOT * DMODEL);
  cvt_f32_f16<<<1024, 256, 0, stream>>>(Wq, wq16, DMODEL * DMODEL);
  cvt_f32_f16<<<1024, 256, 0, stream>>>(Wk, wk16, DMODEL * DMODEL);
  cvt_f32_f16<<<1024, 256, 0, stream>>>(Wv, wv16, DMODEL * DMODEL);
  cvt_f32_f16<<<1024, 256, 0, stream>>>(Wo, wo16, DMODEL * DMODEL);

  // 2) QKV projections (Q pre-scaled by HD^-0.5 = 0.125; commutes with RoPE)
  gemm_qkv<<<1024, 256, 0, stream>>>(x16, wq16, bq, q16, 0.125f);
  gemm_qkv<<<1024, 256, 0, stream>>>(x16, wk16, bk, k16, 1.0f);
  gemm_qkv<<<1024, 256, 0, stream>>>(x16, wv16, bv, v16, 1.0f);

  // 3) RoPE on Q and K (in place; buffers rewritten every call -> deterministic)
  const int npairs = BATCH * NHEADS * SEQ * HDIM / 2;   // 8388608
  rope_f16<<<32768, 256, 0, stream>>>(q16, cosb, sinb, npairs);
  rope_f16<<<32768, 256, 0, stream>>>(k16, cosb, sinb, npairs);

  // 4) fused flash attention (LDS-staged K/V panels)
  attn_flash<<<2048, 256, 0, stream>>>(q16, k16, v16, a16);

  // 5) output projection -> fp32
  gemm_out<<<1024, 256, 0, stream>>>(a16, wo16, bo, out);
}
